// MambaBlock_39144331935827
// MI455X (gfx1250) — compile-verified, matrix operand order checked by reading, not run on hardware
//
#include <hip/hip_runtime.h>
#include <hip/hip_bf16.h>
#include <math.h>

// ---- problem constants (match reference) ----
#define BSZ   2
#define LSEQ  1024
#define DM    1024
#define DI    2048
#define DST   16
#define DTR   128
#define KCONV 4
#define BL    (BSZ * LSEQ)       // 2048 tokens
#define XPN   (DTR + 2 * DST)    // 160

// ---- WMMA / TDM types ----
typedef __attribute__((ext_vector_type(16))) __bf16         v16bf_t;
typedef __attribute__((ext_vector_type(16))) unsigned short v16u16_t;
typedef __attribute__((ext_vector_type(8)))  float          v8f_t;
typedef __attribute__((ext_vector_type(4)))  unsigned int   v4u_t;
typedef __attribute__((ext_vector_type(8)))  int            v8i_t;
typedef __attribute__((ext_vector_type(4)))  int            v4i_t;

#if __has_builtin(__builtin_amdgcn_tensor_load_to_lds) && \
    __has_builtin(__builtin_amdgcn_s_wait_tensorcnt)
#define HAVE_TDM 1
#else
#define HAVE_TDM 0
#endif

__device__ __forceinline__ unsigned short f32_to_bf16(float f) {
  unsigned int u = __float_as_uint(f);
  u += 0x7FFFu + ((u >> 16) & 1u);   // round-to-nearest-even
  return (unsigned short)(u >> 16);
}

// ---------------------------------------------------------------------------
// f32 -> bf16 elementwise (weights + small tensors)
// ---------------------------------------------------------------------------
__global__ void cvt_f32_to_bf16(const float* __restrict__ in,
                                unsigned short* __restrict__ out, int n) {
  int i = blockIdx.x * blockDim.x + threadIdx.x;
  if (i < n) out[i] = f32_to_bf16(in[i]);
}

// ---------------------------------------------------------------------------
// LayerNorm over D_MODEL, emit bf16 activations for the in_proj WMMA GEMM
// ---------------------------------------------------------------------------
__global__ void __launch_bounds__(256) layernorm_bf16(
    const float* __restrict__ x, const float* __restrict__ w,
    const float* __restrict__ b, unsigned short* __restrict__ h) {
  const int t = blockIdx.x;                 // token
  const float* row = x + (size_t)t * DM;
  float s = 0.f, ss = 0.f;
  for (int j = threadIdx.x; j < DM; j += blockDim.x) {
    float v = row[j]; s += v; ss += v * v;
  }
  for (int off = 16; off > 0; off >>= 1) {  // wave32 tree reduce
    s  += __shfl_down(s,  off);
    ss += __shfl_down(ss, off);
  }
  __shared__ float shs[8], shss[8], stats[2];
  const int wave = threadIdx.x >> 5, lane = threadIdx.x & 31;
  if (lane == 0) { shs[wave] = s; shss[wave] = ss; }
  __syncthreads();
  if (threadIdx.x == 0) {
    float S = 0.f, SS = 0.f;
    for (int i = 0; i < (int)(blockDim.x >> 5); ++i) { S += shs[i]; SS += shss[i]; }
    float mu  = S / (float)DM;
    float var = SS / (float)DM - mu * mu;
    stats[0] = mu; stats[1] = rsqrtf(var + 1e-5f);
  }
  __syncthreads();
  const float mu = stats[0], r = stats[1];
  for (int j = threadIdx.x; j < DM; j += blockDim.x) {
    float v = (row[j] - mu) * r * w[j] + b[j];
    h[(size_t)t * DM + j] = f32_to_bf16(v);
  }
}

// ---------------------------------------------------------------------------
// bf16 WMMA GEMM: C[MxN] = A[MxK] * B[KxN] (+ optional f32 residual R)
//   block tile 128(M) x 32(N), K-step 32; 4 waves, each computing 32x32
//   -> 4 v_wmma_f32_16x16x32_bf16 per wave per K-step with A/B frag reuse.
// A tile staged into LDS by the Tensor Data Mover (one tensor_load_to_lds
// per K-step, LDS row padding 64B data + 16B pad = 80B stride), B tile staged
// transposed so both fragments are contiguous ds_load_b128 pairs.
// Requires: M % 128 == 0, N % 32 == 0, K % 32 == 0 (true for all 4 GEMMs).
// ---------------------------------------------------------------------------
#define TMBLK 128
#define TN    32
#define TK    32
#define LDP   40   // padded LDS row stride in ushorts (80 B, 16B-aligned)

__global__ void __launch_bounds__(128) gemm_bf16_wmma(
    const unsigned short* __restrict__ A, const unsigned short* __restrict__ B,
    float* __restrict__ C, const float* __restrict__ R,
    int M, int N, int K) {
  // single shared block so the A-tile sits at LDS offset 0 (TDM destination)
  __shared__ unsigned short lds[TMBLK * LDP + TN * LDP];
#define As(r_, c_)  lds[(r_) * LDP + (c_)]
#define BsT(n_, k_) lds[TMBLK * LDP + (n_) * LDP + (k_)]

  const int tid  = threadIdx.x;
  const int wave = tid >> 5;
  const int lane = tid & 31;
  const int m0   = blockIdx.y * TMBLK;
  const int n0   = blockIdx.x * TN;
  const int mw   = wave * 32;              // wave's 32-row strip
  const int rsel = lane & 15;              // M row / N col within 16x16 tile
  const int half = lane >> 4;              // K-half selector (ISA 16-bit layout)

  v8f_t acc00 = {0.f,0.f,0.f,0.f,0.f,0.f,0.f,0.f};
  v8f_t acc01 = {0.f,0.f,0.f,0.f,0.f,0.f,0.f,0.f};
  v8f_t acc10 = {0.f,0.f,0.f,0.f,0.f,0.f,0.f,0.f};
  v8f_t acc11 = {0.f,0.f,0.f,0.f,0.f,0.f,0.f,0.f};

  for (int k0 = 0; k0 < K; k0 += TK) {
    __syncthreads();   // previous tile fully consumed before overwrite

#if HAVE_TDM
    // ---- A tile via Tensor Data Mover: 32(x) x 128(rows) of 2-byte elems ----
    if (wave == 0) {
      unsigned long long ga =
          (unsigned long long)(A) + ((unsigned long long)(m0)*K + k0) * 2ull;
      v4u_t g0;
      g0[0] = 1u;                                        // count=1, user mode
      g0[1] = 0u;                                        // lds_addr = 0 (As base)
      g0[2] = (unsigned)(ga & 0xFFFFFFFFull);            // global_addr[31:0]
      g0[3] = (unsigned)((ga >> 32) & 0x1FFFFFFu)        // global_addr[56:32]
              | (2u << 30);                              // type = 2 (image)
      v8i_t g1;
      g1[0] = (int)((1u << 16)                           // data_size = 2 bytes
                    | (1u << 20)                         // pad_enable
                    | (3u << 22)                         // pad_interval: 16 DWORDs
                    | (3u << 25));                       // pad_amount: 4 DWORDs
      g1[1] = (int)(((unsigned)K & 0xFFFFu) << 16);      // tensor_dim0[15:0]
      g1[2] = (int)((((unsigned)K >> 16) & 0xFFFFu)      // tensor_dim0[31:16]
                    | (((unsigned)M & 0xFFFFu) << 16));  // tensor_dim1[15:0]
      g1[3] = (int)((((unsigned)M >> 16) & 0xFFFFu)      // tensor_dim1[31:16]
                    | ((unsigned)TK << 16));             // tile_dim0 = 32
      g1[4] = (int)(TMBLK & 0xFFFFu);                    // tile_dim1=128, tile_dim2=0
      g1[5] = (int)(unsigned)K;                          // tensor_dim0_stride[31:0]
      g1[6] = 0;                                         // stride hi, dim1_stride lo
      g1[7] = 0;
      v4i_t gz = {0, 0, 0, 0};
#if __clang_major__ >= 23
      v8i_t gz8 = {0, 0, 0, 0, 0, 0, 0, 0};
      __builtin_amdgcn_tensor_load_to_lds(g0, g1, gz, gz, gz8, 0);
#else
      __builtin_amdgcn_tensor_load_to_lds(g0, g1, gz, gz, 0);
#endif
    }
#else
    // ---- fallback: cooperative A tile load (128x32 bf16, 16B vectors) ----
#pragma unroll
    for (int i = 0; i < 4; ++i) {
      int chunk = tid + i * 128;
      int row = chunk >> 2;
      int col = (chunk & 3) << 3;
      uint4 v = *reinterpret_cast<const uint4*>(A + (size_t)(m0 + row) * K + (k0 + col));
      *reinterpret_cast<uint4*>(&As(row, col)) = v;
    }
#endif

    // ---- B tile: coalesced 16B global load, transposed scatter into LDS ----
    {
      int kr = tid >> 2;              // 0..31
      int n8 = (tid & 3) << 3;        // 0,8,16,24
      union { uint4 v; unsigned short s[8]; } u;
      u.v = *reinterpret_cast<const uint4*>(B + (size_t)(k0 + kr) * N + (n0 + n8));
#pragma unroll
      for (int j = 0; j < 8; ++j) BsT(n8 + j, kr) = u.s[j];
    }
    if (k0 + TK < K) {   // global_prefetch_b8 for the next B tile
      __builtin_prefetch(B + (size_t)(k0 + TK + (tid >> 2)) * N + n0, 0, 1);
    }

#if HAVE_TDM
    if (wave == 0) __builtin_amdgcn_s_wait_tensorcnt(0);
#endif
    __syncthreads();

    // ---- fragments: all contiguous 16B LDS runs (cdna5 §7.12.2 layouts) ----
    v16u16_t af0, af1, bf0, bf1;
    const int r0 = mw + rsel, r1 = mw + 16 + rsel;
#pragma unroll
    for (int e = 0; e < 8; ++e) {
      af0[e]     = As(r0, half * 8 + e);          // K = half*8 + e
      af0[8 + e] = As(r0, 16 + half * 8 + e);     // K = 16 + half*8 + e
      af1[e]     = As(r1, half * 8 + e);
      af1[8 + e] = As(r1, 16 + half * 8 + e);
      bf0[e]     = BsT(rsel,      half * 8 + e);
      bf0[8 + e] = BsT(rsel,      16 + half * 8 + e);
      bf1[e]     = BsT(rsel + 16, half * 8 + e);
      bf1[8 + e] = BsT(rsel + 16, 16 + half * 8 + e);
    }
    acc00 = __builtin_amdgcn_wmma_f32_16x16x32_bf16(
        false, __builtin_bit_cast(v16bf_t, af0), false,
        __builtin_bit_cast(v16bf_t, bf0), (short)0, acc00, false, false);
    acc01 = __builtin_amdgcn_wmma_f32_16x16x32_bf16(
        false, __builtin_bit_cast(v16bf_t, af0), false,
        __builtin_bit_cast(v16bf_t, bf1), (short)0, acc01, false, false);
    acc10 = __builtin_amdgcn_wmma_f32_16x16x32_bf16(
        false, __builtin_bit_cast(v16bf_t, af1), false,
        __builtin_bit_cast(v16bf_t, bf0), (short)0, acc10, false, false);
    acc11 = __builtin_amdgcn_wmma_f32_16x16x32_bf16(
        false, __builtin_bit_cast(v16bf_t, af1), false,
        __builtin_bit_cast(v16bf_t, bf1), (short)0, acc11, false, false);
  }

  // ---- epilogue: C element i <-> M = i + 8*(lane>=16), N = lane&15 ----
  const int nc0 = n0 + rsel;
  const int nc1 = n0 + rsel + 16;
#pragma unroll
  for (int i = 0; i < 8; ++i) {
    int ma = m0 + mw + half * 8 + i;        // A-frag 0 rows
    int mb = ma + 16;                       // A-frag 1 rows
    size_t a0 = (size_t)ma * N + nc0, a1 = (size_t)ma * N + nc1;
    size_t b0 = (size_t)mb * N + nc0, b1 = (size_t)mb * N + nc1;
    float v00 = acc00[i], v01 = acc01[i], v10 = acc10[i], v11 = acc11[i];
    if (R) { v00 += R[a0]; v01 += R[a1]; v10 += R[b0]; v11 += R[b1]; }
    C[a0] = v00; C[a1] = v01; C[b0] = v10; C[b1] = v11;
  }
#undef As
#undef BsT
}

// ---------------------------------------------------------------------------
// Causal depthwise conv (K=4) + SiLU; emit f32 (scan input u) + bf16 (x_proj A)
// ---------------------------------------------------------------------------
__global__ void conv_silu(const float* __restrict__ xr,
                          const float* __restrict__ cw, const float* __restrict__ cb,
                          float* __restrict__ xs, unsigned short* __restrict__ xsb) {
  int idx = blockIdx.x * blockDim.x + threadIdx.x;
  if (idx >= BL * DI) return;
  int t = idx / DI, d = idx % DI;
  int b = t / LSEQ, l = t % LSEQ;
  float o = cb[d];
#pragma unroll
  for (int k = 0; k < KCONV; ++k) {
    int ls = l + k - (KCONV - 1);
    if (ls >= 0)
      o += cw[d * KCONV + k] * xr[(size_t)(b * LSEQ + ls) * (2 * DI) + d];
  }
  o = o / (1.f + __expf(-o));   // SiLU
  xs[idx]  = o;
  xsb[idx] = f32_to_bf16(o);
}

// xp[:, :128] -> bf16 for the dt_proj GEMM
__global__ void extract_dr(const float* __restrict__ xp, unsigned short* __restrict__ dr) {
  int idx = blockIdx.x * blockDim.x + threadIdx.x;
  if (idx >= BL * DTR) return;
  int t = idx / DTR, j = idx % DTR;
  dr[idx] = f32_to_bf16(xp[(size_t)t * XPN + j]);
}

// delta = softplus(delta_raw + dt_proj_b), in place
__global__ void softplus_bias(float* __restrict__ dl, const float* __restrict__ bias) {
  int idx = blockIdx.x * blockDim.x + threadIdx.x;
  if (idx >= BL * DI) return;
  float v = dl[idx] + bias[idx % DI];
  dl[idx] = (v > 20.f) ? v : log1pf(__expf(v));
}

// ---------------------------------------------------------------------------
// Selective scan: one wave = two (b,d) channels; 16 lanes = 16 states.
// ---------------------------------------------------------------------------
__global__ void __launch_bounds__(256) selective_scan(
    const float* __restrict__ delta, const float* __restrict__ xs,
    const float* __restrict__ xp, const float* __restrict__ A_log,
    const float* __restrict__ Dp, float* __restrict__ y) {
  const int lane = threadIdx.x & 31;
  const int wave = threadIdx.x >> 5;
  const int n  = lane & 15;
  const int ch = ((blockIdx.x * 8 + wave) << 1) + (lane >> 4);  // (b,d) channel
  const int b = ch / DI, d = ch % DI;
  const float a     = -__expf(A_log[d * DST + n]);
  const float dskip = Dp[d];
  float h = 0.f;
  for (int l = 0; l < LSEQ; ++l) {
    const size_t t  = (size_t)b * LSEQ + l;
    const float dt = delta[t * DI + d];
    const float u  = xs[t * DI + d];
    const float Bv = xp[t * XPN + DTR + n];
    const float Cv = xp[t * XPN + DTR + DST + n];
    h = __expf(dt * a) * h + dt * Bv * u;
    float p = h * Cv;
    p += __shfl_xor(p, 8);
    p += __shfl_xor(p, 4);
    p += __shfl_xor(p, 2);
    p += __shfl_xor(p, 1);
    if (n == 0) y[t * DI + d] = p + u * dskip;
  }
}

// g = y * silu(res), emitted bf16 for the out_proj GEMM
__global__ void gate_silu(const float* __restrict__ y, const float* __restrict__ xr,
                          unsigned short* __restrict__ g) {
  int idx = blockIdx.x * blockDim.x + threadIdx.x;
  if (idx >= BL * DI) return;
  int t = idx / DI, d = idx % DI;
  float r = xr[(size_t)t * (2 * DI) + DI + d];
  float v = y[idx] * (r / (1.f + __expf(-r)));
  g[idx] = f32_to_bf16(v);
}

// ---------------------------------------------------------------------------
extern "C" void kernel_launch(void* const* d_in, const int* in_sizes, int n_in,
                              void* d_out, int out_size, void* d_ws, size_t ws_size,
                              hipStream_t stream) {
  const float* x        = (const float*)d_in[0];
  const float* norm_w   = (const float*)d_in[1];
  const float* norm_b   = (const float*)d_in[2];
  const float* in_proj  = (const float*)d_in[3];
  const float* conv_w   = (const float*)d_in[4];
  const float* conv_b   = (const float*)d_in[5];
  const float* x_proj   = (const float*)d_in[6];
  const float* dt_proj  = (const float*)d_in[7];
  const float* dt_b     = (const float*)d_in[8];
  const float* A_log    = (const float*)d_in[9];
  const float* Dp       = (const float*)d_in[10];
  const float* out_proj = (const float*)d_in[11];
  float* out = (float*)d_out;

  char* ws = (char*)d_ws;
  size_t off = 0;
  auto alloc = [&](size_t bytes) -> void* {
    void* p = ws + off;
    off = (off + bytes + 255) & ~(size_t)255;
    return p;
  };
  unsigned short* w_in_b  = (unsigned short*)alloc((size_t)DM * 2 * DI * 2);
  unsigned short* w_xp_b  = (unsigned short*)alloc((size_t)DI * XPN * 2);
  unsigned short* w_dt_b  = (unsigned short*)alloc((size_t)DTR * DI * 2);
  unsigned short* w_out_b = (unsigned short*)alloc((size_t)DI * DM * 2);
  unsigned short* h_b     = (unsigned short*)alloc((size_t)BL * DM * 2);
  float*          xr      = (float*)alloc((size_t)BL * 2 * DI * 4);
  float*          xs      = (float*)alloc((size_t)BL * DI * 4);
  unsigned short* xs_b    = (unsigned short*)alloc((size_t)BL * DI * 2);
  float*          xp      = (float*)alloc((size_t)BL * XPN * 4);
  unsigned short* dr_b    = (unsigned short*)alloc((size_t)BL * DTR * 2);
  float*          delta   = (float*)alloc((size_t)BL * DI * 4);
  float*          y       = (float*)alloc((size_t)BL * DI * 4);
  unsigned short* g_b     = (unsigned short*)alloc((size_t)BL * DI * 2);

  const int thr = 256;
  // weights -> bf16 (deterministic, once per launch)
  cvt_f32_to_bf16<<<(DM * 2 * DI + thr - 1) / thr, thr, 0, stream>>>(in_proj, w_in_b, DM * 2 * DI);
  cvt_f32_to_bf16<<<(DI * XPN + thr - 1) / thr, thr, 0, stream>>>(x_proj, w_xp_b, DI * XPN);
  cvt_f32_to_bf16<<<(DTR * DI + thr - 1) / thr, thr, 0, stream>>>(dt_proj, w_dt_b, DTR * DI);
  cvt_f32_to_bf16<<<(DI * DM + thr - 1) / thr, thr, 0, stream>>>(out_proj, w_out_b, DI * DM);

  // 1. LayerNorm
  layernorm_bf16<<<BL, 256, 0, stream>>>(x, norm_w, norm_b, h_b);
  // 2. in_proj: [2048x1024] @ [1024x4096]
  gemm_bf16_wmma<<<dim3(2 * DI / TN, BL / TMBLK), 128, 0, stream>>>(
      h_b, w_in_b, xr, nullptr, BL, 2 * DI, DM);
  // 3. causal depthwise conv + SiLU
  conv_silu<<<(BL * DI + thr - 1) / thr, thr, 0, stream>>>(xr, conv_w, conv_b, xs, xs_b);
  // 4. x_proj: [2048x2048] @ [2048x160]
  gemm_bf16_wmma<<<dim3(XPN / TN, BL / TMBLK), 128, 0, stream>>>(
      xs_b, w_xp_b, xp, nullptr, BL, XPN, DI);
  // 5. dt_proj: [2048x128] @ [128x2048]
  extract_dr<<<(BL * DTR + thr - 1) / thr, thr, 0, stream>>>(xp, dr_b);
  gemm_bf16_wmma<<<dim3(DI / TN, BL / TMBLK), 128, 0, stream>>>(
      dr_b, w_dt_b, delta, nullptr, BL, DI, DTR);
  softplus_bias<<<(BL * DI + thr - 1) / thr, thr, 0, stream>>>(delta, dt_b);
  // 6. selective scan (4096 channels, 2 per wave, 8 waves/block -> 256 blocks)
  selective_scan<<<256, 256, 0, stream>>>(delta, xs, xp, A_log, Dp, y);
  // 7. gate with silu(res)
  gate_silu<<<(BL * DI + thr - 1) / thr, thr, 0, stream>>>(y, xr, g_b);
  // 8. out_proj with fused residual add: [2048x2048] @ [2048x1024] + x
  gemm_bf16_wmma<<<dim3(DM / TN, BL / TMBLK), 128, 0, stream>>>(
      g_b, w_out_b, out, x, BL, DM, DI);
}
